// CustomRGCN_71098888618668
// MI455X (gfx1250) — matrix operand order repaired and checked by reading, not compiled, not run on hardware
//
#include <hip/hip_runtime.h>
#include <hip/hip_bf16.h>

// ---------------------------------------------------------------------------
// Types for CDNA5 WMMA (gfx1250, wave32)
// ---------------------------------------------------------------------------
typedef __attribute__((ext_vector_type(16))) __bf16 v16bf;
typedef __attribute__((ext_vector_type(8)))  float  v8f;

union V16BF {
    v16bf v;
    unsigned short s[16];
    unsigned int   u[8];
    uint4          q[2];
};

static __device__ __forceinline__ unsigned short f2bf(float f) {
    unsigned int u = __float_as_uint(f);
    unsigned int r = (u + 0x7FFFu + ((u >> 16) & 1u)) >> 16;   // RNE bf16
    return (unsigned short)r;
}

static __device__ __forceinline__ float lrelu(float v) {
    return v > 0.0f ? v : 0.01f * v;
}

// ---------------------------------------------------------------------------
// bf16 GEMM with v_wmma_f32_16x16x32_bf16, pre-swizzled B operand and
// ping-pong operand loads. One wave -> 16x64 output tile.
// A[M,K] row-major bf16; Bp packed [K/32][Ncols/16][32 lanes][16 bf16].
// Requirements: M%16==0, K%64==0, Ncols%64==0.
// ---------------------------------------------------------------------------
__global__ __launch_bounds__(256)
void gemm_bf16_wmma(const unsigned short* __restrict__ A, int lda,
                    const V16BF* __restrict__ Bp,
                    const float* __restrict__ bias,
                    float* __restrict__ C, int ldc,
                    int M, int K, int Ncols, int act)
{
    const int tilesN64 = Ncols >> 6;
    const int tilesN16 = Ncols >> 4;
    const int tilesM   = M >> 4;
    int wg = blockIdx.x * 8 + (threadIdx.x >> 5);
    if (wg >= tilesM * tilesN64) return;            // whole-wave exit, EXEC stays full
    const int tm   = wg / tilesN64;
    const int tn   = wg % tilesN64;
    const int lane = threadIdx.x & 31;
    const int half = lane >> 4;
    const int l15  = lane & 15;

    const size_t rowA = (size_t)(tm * 16 + l15);
    const unsigned int* ApU = (const unsigned int*)(A + rowA * (size_t)lda) + 8 * half;
    const V16BF* Bbase = Bp + ((size_t)tn << 2) * 32 + lane;
    const size_t bStride = (size_t)tilesN16 * 32;

    auto loadA = [&](int kblk, V16BF& a) {
        const uint4* p = (const uint4*)(ApU + (kblk << 4));
        a.q[0] = p[0];
        a.q[1] = p[1];
    };
    auto loadB = [&](int kblk, V16BF* b) {
        const V16BF* br = Bbase + (size_t)kblk * bStride;
        b[0] = br[0];
        b[1] = br[32];
        b[2] = br[64];
        b[3] = br[96];
    };

    v8f acc0 = {0.f,0.f,0.f,0.f,0.f,0.f,0.f,0.f};
    v8f acc[4];
    acc[0] = acc0; acc[1] = acc0; acc[2] = acc0; acc[3] = acc0;

    const int nKblk = K >> 5;                // even for all GEMMs used here
    V16BF aP, aQ;
    V16BF bP[4], bQ[4];
    loadA(0, aP); loadB(0, bP);

    for (int kblk = 0; kblk < nKblk; kblk += 2) {
        loadA(kblk + 1, aQ); loadB(kblk + 1, bQ);
#pragma unroll
        for (int s = 0; s < 4; ++s)
            acc[s] = __builtin_amdgcn_wmma_f32_16x16x32_bf16(
                false, aP.v, false, bP[s].v, (short)0, acc[s], false, false);

        const int kn = (kblk + 2 < nKblk) ? (kblk + 2) : kblk;
        loadA(kn, aP); loadB(kn, bP);
#pragma unroll
        for (int s = 0; s < 4; ++s)
            acc[s] = __builtin_amdgcn_wmma_f32_16x16x32_bf16(
                false, aQ.v, false, bQ[s].v, (short)0, acc[s], false, false);
    }

#pragma unroll
    for (int s = 0; s < 4; ++s) {
        const int col = (tn << 6) + (s << 4) + l15;
        const float bi = bias ? bias[col] : 0.0f;
#pragma unroll
        for (int v = 0; v < 8; ++v) {
            const int row = tm * 16 + v + 8 * half;
            float val = acc[s][v] + bi;
            if (act) val = lrelu(val);
            C[(size_t)row * ldc + col] = val;
        }
    }
}

// ---------------------------------------------------------------------------
// Operand packing kernels
// ---------------------------------------------------------------------------
__global__ void k_f32_to_bf16(const float* __restrict__ in,
                              unsigned short* __restrict__ out, long long n) {
    long long i = (long long)blockIdx.x * blockDim.x + threadIdx.x;
    if (i < n) out[i] = f2bf(in[i]);
}

__global__ void k_pack_b(const float* __restrict__ w,
                         unsigned short* __restrict__ out,
                         int K, int Ncols) {
    long long total = (long long)K * Ncols;
    long long idx = (long long)blockIdx.x * blockDim.x + threadIdx.x;
    if (idx >= total) return;
    int j        = (int)(idx & 15);
    long long t  = idx >> 4;
    int lane     = (int)(t & 31);
    t >>= 5;
    int tilesN16 = Ncols >> 4;
    int tn       = (int)(t % tilesN16);
    int kblk     = (int)(t / tilesN16);
    int half     = lane >> 4;
    int n        = tn * 16 + (lane & 15);
    int k        = kblk * 32 + 16 * half + 2 * (j >> 1) + (j & 1);
    out[idx] = f2bf(w[(size_t)k * Ncols + n]);
}

__global__ void k_pack_rel_b(const float* __restrict__ w_rel,
                             const float* __restrict__ w_root,
                             unsigned short* __restrict__ out, int D) {
    const int Ncols = 4 * D;
    long long total = (long long)D * Ncols;
    long long idx = (long long)blockIdx.x * blockDim.x + threadIdx.x;
    if (idx >= total) return;
    int j        = (int)(idx & 15);
    long long t  = idx >> 4;
    int lane     = (int)(t & 31);
    t >>= 5;
    int tilesN16 = Ncols >> 4;
    int tn       = (int)(t % tilesN16);
    int kblk     = (int)(t / tilesN16);
    int half     = lane >> 4;
    int n        = tn * 16 + (lane & 15);
    int k        = kblk * 32 + 16 * half + 2 * (j >> 1) + (j & 1);
    int r        = n / D;
    int nn       = n % D;
    float f = (r < 3) ? w_rel[((size_t)r * D + k) * D + nn]
                      : w_root[(size_t)k * D + nn];
    out[idx] = f2bf(f);
}

// ---------------------------------------------------------------------------
// CSR-by-(dst,rel) construction (built once; graph static across layers)
// ---------------------------------------------------------------------------
__global__ void k_edge_count(const int* __restrict__ dst,
                             const int* __restrict__ et,
                             int* __restrict__ cnt, int E) {
    int e = blockIdx.x * blockDim.x + threadIdx.x;
    if (e < E) atomicAdd(&cnt[(size_t)dst[e] * 3 + et[e]], 1);
}

// 1024-element block inclusive scan; writes per-block totals
__global__ __launch_bounds__(256)
void k_scan_pass1(const int* __restrict__ in, int* __restrict__ incl,
                  int* __restrict__ bsum, int n) {
    __shared__ int sh[1024];
    int base = blockIdx.x * 1024;
    for (int i = threadIdx.x; i < 1024; i += 256)
        sh[i] = (base + i < n) ? in[base + i] : 0;
    __syncthreads();
    for (int offs = 1; offs < 1024; offs <<= 1) {
        int vals[4];
        for (int i = threadIdx.x, j = 0; i < 1024; i += 256, ++j)
            vals[j] = (i >= offs) ? sh[i - offs] : 0;
        __syncthreads();
        for (int i = threadIdx.x, j = 0; i < 1024; i += 256, ++j)
            sh[i] += vals[j];
        __syncthreads();
    }
    for (int i = threadIdx.x; i < 1024; i += 256)
        if (base + i < n) incl[base + i] = sh[i];
    if (threadIdx.x == 0) bsum[blockIdx.x] = sh[1023];
}

__global__ void k_scan_pass2(int* __restrict__ bsum, int nb) {
    if (blockIdx.x == 0 && threadIdx.x == 0) {
        int acc = 0;
        for (int i = 0; i < nb; ++i) { int v = bsum[i]; bsum[i] = acc; acc += v; }
    }
}

// offsets[i+1] = incl[i] + bsum[i/1024]; offsets[0] = 0
__global__ void k_scan_pass3(const int* __restrict__ incl,
                             const int* __restrict__ bsum,
                             int* __restrict__ offsets, int n) {
    int i = blockIdx.x * blockDim.x + threadIdx.x;
    if (i < n) offsets[i + 1] = incl[i] + bsum[i >> 10];
    if (i == 0) offsets[0] = 0;
}

__global__ void k_edge_place(const int* __restrict__ dst,
                             const int* __restrict__ et,
                             int* __restrict__ run,
                             int* __restrict__ eidx, int E) {
    int e = blockIdx.x * blockDim.x + threadIdx.x;
    if (e >= E) return;
    int key = dst[e] * 3 + et[e];
    int pos = atomicAdd(&run[key], 1);
    eidx[pos] = e;
}

// ---------------------------------------------------------------------------
// RGCN aggregate (fused mean-per-relation + root + bias), wave per node.
// out[n,:] = sum_r mean_{e in seg(n,r)} y[src(e), r*320:*] + y[n,960:1280] + bias
// ---------------------------------------------------------------------------
__global__ __launch_bounds__(256)
void k_rgcn_aggregate(const float* __restrict__ y,
                      const int* __restrict__ src,
                      const int* __restrict__ eidx,
                      const int* __restrict__ offsets,
                      const float* __restrict__ bias,
                      float* __restrict__ out, int N) {
    int n = blockIdx.x * 8 + (threadIdx.x >> 5);    // one wave per node
    if (n >= N) return;
    int lane = threadIdx.x & 31;
    float acc[10];
    const float* yr = y + (size_t)n * 1280 + 960 + lane;
#pragma unroll
    for (int i = 0; i < 10; ++i) acc[i] = yr[i * 32] + bias[lane + i * 32];
    for (int r = 0; r < 3; ++r) {
        int seg = n * 3 + r;                        // uniform per wave
        int start = offsets[seg], end = offsets[seg + 1];
        if (end <= start) continue;                 // uniform branch
        float tmp[10];
#pragma unroll
        for (int i = 0; i < 10; ++i) tmp[i] = 0.f;
        for (int p = start; p < end; ++p) {
            const float* yp = y + (size_t)src[eidx[p]] * 1280 + r * 320 + lane;
#pragma unroll
            for (int i = 0; i < 10; ++i) tmp[i] += yp[i * 32];
        }
        float inv = 1.0f / (float)(end - start);
#pragma unroll
        for (int i = 0; i < 10; ++i) acc[i] += tmp[i] * inv;
    }
    float* op = out + (size_t)n * 320 + lane;
#pragma unroll
    for (int i = 0; i < 10; ++i) op[i * 32] = acc[i];
}

// ---------------------------------------------------------------------------
// Misc kernels
// ---------------------------------------------------------------------------
__global__ void k_small_proj(const float* __restrict__ num,
                             const float* __restrict__ cat,
                             const float* __restrict__ wn, const float* __restrict__ bn_,
                             const float* __restrict__ wc, const float* __restrict__ bc,
                             float* __restrict__ h, int N) {
    long long idx = (long long)blockIdx.x * blockDim.x + threadIdx.x;
    int n = (int)(idx >> 7);
    int c = (int)(idx & 127);
    if (n >= N) return;
    if (c < 64) {
        float acc = bn_[c];
        for (int k = 0; k < 5; ++k) acc += num[(size_t)n * 5 + k] * wn[k * 64 + c];
        h[(size_t)n * 320 + c] = lrelu(acc);
    } else {
        int cc = c - 64;
        float acc = bc[cc];
        for (int k = 0; k < 3; ++k) acc += cat[(size_t)n * 3 + k] * wc[k * 64 + cc];
        h[(size_t)n * 320 + 64 + cc] = lrelu(acc);
    }
}

__global__ __launch_bounds__(256)
void k_bn_stats(const float* __restrict__ x, float* __restrict__ stats,
                int N, int D) {
    const int c = blockIdx.x;
    float s = 0.f, s2 = 0.f;
    for (int n = threadIdx.x; n < N; n += 256) {
        float v = x[(size_t)n * D + c];
        s += v; s2 += v * v;
    }
    __shared__ float sh[256], sh2[256];
    sh[threadIdx.x] = s; sh2[threadIdx.x] = s2;
    __syncthreads();
    for (int st = 128; st > 0; st >>= 1) {
        if (threadIdx.x < st) {
            sh[threadIdx.x]  += sh[threadIdx.x + st];
            sh2[threadIdx.x] += sh2[threadIdx.x + st];
        }
        __syncthreads();
    }
    if (threadIdx.x == 0) {
        float m = sh[0] / (float)N;
        float var = sh2[0] / (float)N - m * m;
        stats[c]     = m;
        stats[D + c] = rsqrtf(var + 1e-5f);
    }
}

__global__ void k_bn_apply(const float* __restrict__ x,
                           const float* __restrict__ stats,
                           const float* __restrict__ gamma,
                           const float* __restrict__ beta,
                           float* __restrict__ xout,
                           unsigned short* __restrict__ xb, int N, int D) {
    long long idx = (long long)blockIdx.x * blockDim.x + threadIdx.x;
    if (idx >= (long long)N * D) return;
    int c = (int)(idx % D);
    float v = (x[idx] - stats[c]) * stats[D + c] * gamma[c] + beta[c];
    v = lrelu(v);
    xout[idx] = v;
    xb[idx]   = f2bf(v);
}

__global__ void k_final_fc(const float* __restrict__ x,
                           const float* __restrict__ w,
                           const float* __restrict__ b,
                           float* __restrict__ out, int N) {
    int n = blockIdx.x * blockDim.x + threadIdx.x;
    if (n >= N) return;
    float a0 = b[0], a1 = b[1];
    const float* xr = x + (size_t)n * 320;
    for (int k = 0; k < 320; ++k) {
        float v = xr[k];
        a0 += v * w[k * 2 + 0];
        a1 += v * w[k * 2 + 1];
    }
    out[(size_t)n * 2 + 0] = a0;
    out[(size_t)n * 2 + 1] = a1;
}

// ---------------------------------------------------------------------------
// Orchestration
// ---------------------------------------------------------------------------
static inline int cdiv_ll(long long a, long long b) { return (int)((a + b - 1) / b); }

extern "C" void kernel_launch(void* const* d_in, const int* in_sizes, int n_in,
                              void* d_out, int out_size, void* d_ws, size_t ws_size,
                              hipStream_t stream) {
    const int D = 320;
    const int N = in_sizes[0] / 5;    // num_prop is [N,5]
    const int E = in_sizes[6];        // edge_type is [E]
    const int nSeg = N * 3;

    const float* num_prop  = (const float*)d_in[0];
    const float* cat_prop  = (const float*)d_in[1];
    const float* tweet_emb = (const float*)d_in[2];
    const float* user_emb  = (const float*)d_in[3];
    const float* uname_emb = (const float*)d_in[4];
    const int*   edge_idx  = (const int*)d_in[5];
    const int*   edge_type = (const int*)d_in[6];
    const float* w_num   = (const float*)d_in[7];   const float* b_num   = (const float*)d_in[8];
    const float* w_cat   = (const float*)d_in[9];   const float* b_cat   = (const float*)d_in[10];
    const float* w_tweet = (const float*)d_in[11];  const float* b_tweet = (const float*)d_in[12];
    const float* w_user  = (const float*)d_in[13];  const float* b_user  = (const float*)d_in[14];
    const float* w_uname = (const float*)d_in[15];  const float* b_uname = (const float*)d_in[16];
    const float* w_in    = (const float*)d_in[17];  const float* b_in    = (const float*)d_in[18];
    const float* w_rel1  = (const float*)d_in[19];  const float* w_root1 = (const float*)d_in[20];
    const float* b_rgcn1 = (const float*)d_in[21];
    const float* gamma1  = (const float*)d_in[22];  const float* beta1   = (const float*)d_in[23];
    const float* w_rel2  = (const float*)d_in[24];  const float* w_root2 = (const float*)d_in[25];
    const float* b_rgcn2 = (const float*)d_in[26];
    const float* gamma2  = (const float*)d_in[27];  const float* beta2   = (const float*)d_in[28];
    const float* w_o1    = (const float*)d_in[29];  const float* b_o1    = (const float*)d_in[30];
    const float* w_o2    = (const float*)d_in[31];  const float* b_o2    = (const float*)d_in[32];
    const float* w_o3    = (const float*)d_in[33];  const float* b_o3    = (const float*)d_in[34];

    const int* src = edge_idx;
    const int* dst = edge_idx + E;

    // workspace layout
    char* ws = (char*)d_ws;
    size_t off = 0;
    auto alloc = [&](size_t bytes) -> void* {
        void* p = ws + off;
        off += (bytes + 255) & ~(size_t)255;
        return p;
    };
    unsigned short* xb    = (unsigned short*)alloc((size_t)N * D * 2);
    float*          x     = (float*)alloc((size_t)N * D * 4);
    float*          x2    = (float*)alloc((size_t)N * D * 4);
    float*          y     = (float*)alloc((size_t)N * 4 * D * 4);
    unsigned short* wb    = (unsigned short*)alloc((size_t)2 << 20);
    float*          stats = (float*)alloc((size_t)2 * D * 4);
    int*            cnt_i = (int*)alloc((size_t)nSeg * 4);
    int*            incl  = (int*)alloc((size_t)nSeg * 4);
    int*            bsum  = (int*)alloc((size_t)1024 * 4);
    int*            offs  = (int*)alloc(((size_t)nSeg + 1) * 4);
    int*            run   = (int*)alloc((size_t)nSeg * 4);
    int*            eidx  = (int*)alloc((size_t)E * 4);

    unsigned short* emb_bf = (unsigned short*)y;  // reuse y pre-layer

    const int TB = 256;
    auto launch_gemm = [&](const unsigned short* A, int lda,
                           const unsigned short* Bp,
                           const float* bias, float* C, int ldc,
                           int M, int K, int Ncols, int act) {
        long long waves = (long long)(M / 16) * (Ncols / 64);
        gemm_bf16_wmma<<<cdiv_ll(waves, 8), TB, 0, stream>>>(
            A, lda, (const V16BF*)Bp, bias, C, ldc, M, K, Ncols, act);
    };
    auto to_bf16 = [&](const float* in, unsigned short* out, long long n) {
        k_f32_to_bf16<<<cdiv_ll(n, TB), TB, 0, stream>>>(in, out, n);
    };
    auto pack_b = [&](const float* w, unsigned short* out, int K, int Ncols) {
        k_pack_b<<<cdiv_ll((long long)K * Ncols, TB), TB, 0, stream>>>(w, out, K, Ncols);
    };

    // ---- Build CSR-by-(dst,rel) once (graph static across layers) -----------
    hipMemsetAsync(cnt_i, 0, (size_t)nSeg * 4, stream);
    k_edge_count<<<cdiv_ll(E, TB), TB, 0, stream>>>(dst, edge_type, cnt_i, E);
    const int nScanBlk = cdiv_ll(nSeg, 1024);
    k_scan_pass1<<<nScanBlk, 256, 0, stream>>>(cnt_i, incl, bsum, nSeg);
    k_scan_pass2<<<1, 32, 0, stream>>>(bsum, nScanBlk);
    k_scan_pass3<<<cdiv_ll(nSeg, TB), TB, 0, stream>>>(incl, bsum, offs, nSeg);
    hipMemcpyAsync(run, offs, (size_t)nSeg * 4, hipMemcpyDeviceToDevice, stream);
    k_edge_place<<<cdiv_ll(E, TB), TB, 0, stream>>>(dst, edge_type, run, eidx, E);

    // ---- Stage 0: input projections -> h (in x2), [N,320] -------------------
    k_small_proj<<<cdiv_ll((long long)N * 128, TB), TB, 0, stream>>>(
        num_prop, cat_prop, w_num, b_num, w_cat, b_cat, x2, N);

    const float* embs[3]  = {tweet_emb, user_emb, uname_emb};
    const float* wps[3]   = {w_tweet, w_user, w_uname};
    const float* bps[3]   = {b_tweet, b_user, b_uname};
    const int colOff[3]   = {128, 192, 256};
    for (int i = 0; i < 3; ++i) {
        pack_b(wps[i], wb, 768, 64);
        to_bf16(embs[i], emb_bf, (long long)N * 768);
        launch_gemm(emb_bf, 768, wb, bps[i], x2 + colOff[i], 320, N, 768, 64, 1);
    }

    // ---- Stage 1: x = lrelu(h @ w_in + b_in) --------------------------------
    to_bf16(x2, xb, (long long)N * D);
    pack_b(w_in, wb, D, D);
    launch_gemm(xb, D, wb, b_in, x, D, N, D, D, 1);
    to_bf16(x, xb, (long long)N * D);

    // ---- RGCN layers --------------------------------------------------------
    const float* wrel[2]  = {w_rel1, w_rel2};
    const float* wroot[2] = {w_root1, w_root2};
    const float* brg[2]   = {b_rgcn1, b_rgcn2};
    const float* gam[2]   = {gamma1, gamma2};
    const float* bet[2]   = {beta1, beta2};
    for (int l = 0; l < 2; ++l) {
        k_pack_rel_b<<<cdiv_ll((long long)D * 4 * D, TB), TB, 0, stream>>>(
            wrel[l], wroot[l], wb, D);
        launch_gemm(xb, D, wb, nullptr, y, 4 * D, N, D, 4 * D, 0);
        k_rgcn_aggregate<<<cdiv_ll(N, 8), TB, 0, stream>>>(
            y, src, eidx, offs, brg[l], x2, N);
        k_bn_stats<<<D, 256, 0, stream>>>(x2, stats, N, D);
        k_bn_apply<<<cdiv_ll((long long)N * D, TB), TB, 0, stream>>>(
            x2, stats, gam[l], bet[l], x, xb, N, D);
    }

    // ---- Output MLP ---------------------------------------------------------
    pack_b(w_o1, wb, D, D);
    launch_gemm(xb, D, wb, b_o1, x2, D, N, D, D, 1);
    to_bf16(x2, xb, (long long)N * D);
    pack_b(w_o2, wb, D, D);
    launch_gemm(xb, D, wb, b_o2, x, D, N, D, D, 1);

    k_final_fc<<<cdiv_ll(N, TB), TB, 0, stream>>>(x, w_o3, b_o3, (float*)d_out, N);
}
